// SentenceEmbedding_34643206209935
// MI455X (gfx1250) — compile-verified, matrix operand order checked by reading, not run on hardware
//
#include <hip/hip_runtime.h>
#include <hip/hip_bf16.h>
#include <math.h>

// Problem constants (reference: S,B,E,H1,H2 = 512,256,300,16,64)
#define S_LEN 512
#define B_SZ  256

typedef __attribute__((ext_vector_type(16))) _Float16 v16h;
typedef __attribute__((ext_vector_type(8)))  float    v8f;

union FragH { v16h v; _Float16 h[16]; uint4 q[2]; };
union FragF { v8f  v; float    f[8];  };

// Load one 16x32 f16 WMMA operand fragment for this lane.
// Per ISA 7.12.2: lane l -> row (l&15); halfwords at K = k0..k0+7 and k0+16..k0+23,
// where k0 = kc*32 + ((l>=16)?8:0). Both chunks are 16B-aligned contiguous runs.
__device__ inline v16h ld_frag(const _Float16* row, int k0) {
  FragH a;
  a.q[0] = *(const uint4*)(row + k0);
  a.q[1] = *(const uint4*)(row + k0 + 16);
  return a.v;
}

__device__ inline float sigm(float x) { return 1.0f / (1.0f + __expf(-x)); }

// ---------------------------------------------------------------------------
// fp32 (rows x cols) -> f16 (rows x kpad), zero K-padding. Branch-free per row.
// Used for all weights AND for x itself (rows = S*B), so every GEMM operand
// is b128-loadable with no bounds checks.
// ---------------------------------------------------------------------------
__global__ void cvt_pad_kernel(const float* __restrict__ src, _Float16* __restrict__ dst,
                               int rows, int cols, int kpad) {
  int i = blockIdx.x * blockDim.x + threadIdx.x;
  if (i >= rows * kpad) return;
  int r = i / kpad, c = i % kpad;
  dst[i] = (c < cols) ? (_Float16)src[(size_t)r * cols + c] : (_Float16)0.f;
}

// ---------------------------------------------------------------------------
// xp = A @ W^T + bih.  A: (SB x kpad) f16 K-padded.  W: f16 (N x kpad) row-major
// -> B[k][n] = W[n][k].  One wave computes NT adjacent 16x16 output tiles,
// loading each A fragment once per K-chunk and reusing it across NT WMMAs.
// ---------------------------------------------------------------------------
template<int NT>
__global__ __launch_bounds__(32)
void xproj_kernel(const _Float16* __restrict__ A, int kpad,
                  const _Float16* __restrict__ W, const float* __restrict__ bih,
                  float* __restrict__ XP, int N, int SB) {
  const int lane = threadIdx.x & 31;
  const int mlo  = lane & 15;
  const int hi   = lane >> 4;
  const int ko   = hi * 8;
  const int r0   = blockIdx.x * 16;
  const int n0   = blockIdx.y * (16 * NT);

  FragF c[NT];
#pragma unroll
  for (int tt = 0; tt < NT; ++tt) {
    float bv = bih[n0 + tt * 16 + mlo];     // C/D layout: lane owns column (l&15)
#pragma unroll
    for (int g = 0; g < 8; ++g) c[tt].f[g] = bv;
  }

  const _Float16* Arow = A + (size_t)(r0 + mlo) * kpad;
  const int kiters = kpad >> 5;
  for (int kc = 0; kc < kiters; ++kc) {
    v16h a = ld_frag(Arow, kc * 32 + ko);
#pragma unroll
    for (int tt = 0; tt < NT; ++tt) {
      const _Float16* Brow = W + (size_t)(n0 + tt * 16 + mlo) * kpad;
      v16h b = ld_frag(Brow, kc * 32 + ko);
      c[tt].v = __builtin_amdgcn_wmma_f32_16x16x32_f16(false, a, false, b,
                                                       (short)0, c[tt].v, false, false);
    }
  }

#pragma unroll
  for (int tt = 0; tt < NT; ++tt)
#pragma unroll
    for (int g = 0; g < 8; ++g)
      XP[(size_t)(r0 + g + hi * 8) * N + n0 + tt * 16 + mlo] = c[tt].f[g];
}

// ---------------------------------------------------------------------------
// Sequential GRU scan. One block per 16 batch rows; whh + h state in LDS.
// Per step: hp = h@whh^T + bhh via WMMA (one 16x16 N-tile per wave), gates in
// fp32. The xp gate inputs for step t+1 are staged into an LDS double buffer
// with CDNA5 async-to-LDS loads (ASYNCcnt), overlapping the step-t compute.
// ---------------------------------------------------------------------------
template<int H>
__global__ __launch_bounds__((3 * H / 16) * 32)
void gru_scan_kernel(const float* __restrict__ XP,          // (S*B x 3H) fp32
                     const _Float16* __restrict__ Whh,      // (3H x HP) f16, K-padded
                     const float* __restrict__ bhh,         // (3H) fp32
                     const float* __restrict__ h0,          // (B x H) fp32 or null
                     _Float16* __restrict__ Yout,           // (S*B x HP) f16 or null
                     float* __restrict__ hT,                // (B x H) fp32 or null
                     int S, int B) {
  constexpr int HP   = (H == 16) ? 32 : 64;   // K padded to WMMA K=32 granule
  constexpr int N3   = 3 * H;
  constexpr int NT   = N3 / 16;               // 3 or 12 N-tiles
  constexpr int TPB  = NT * 32;               // one wave per N-tile
  constexpr int KCN  = HP / 32;               // K chunks per WMMA chain
  constexpr int CHNK = (16 * N3) / 4;         // float4 chunks per xp tile
  constexpr int CPT  = CHNK / TPB;            // async b128 loads per thread (= 2)

  __shared__ __align__(16) _Float16 sWhh[N3 * HP];
  __shared__ __align__(16) _Float16 sHh[16 * HP];   // h in f16, zero K-pad
  __shared__ __align__(16) float sXp[2][16 * N3];   // async-staged gate inputs
  __shared__ float sH [16 * H];
  __shared__ float sHp[16 * N3];
  __shared__ float sBhh[N3];

  const int tid = threadIdx.x;
  const int bt0 = blockIdx.x * 16;

  for (int i = tid; i < N3 * HP; i += TPB) sWhh[i] = Whh[i];
  for (int i = tid; i < N3;      i += TPB) sBhh[i] = bhh[i];
  for (int i = tid; i < 16 * H;  i += TPB)
    sH[i] = h0 ? h0[(size_t)(bt0 + i / H) * H + (i % H)] : 0.0f;
  __syncthreads();
  for (int i = tid; i < 16 * HP; i += TPB) {
    int b = i / HP, j = i % HP;
    sHh[i] = (j < H) ? (_Float16)sH[b * H + j] : (_Float16)0.f;
  }

  const int wave = tid >> 5;
  const int lane = tid & 31;
  const int mlo  = lane & 15;
  const int hi   = lane >> 4;
  const int ko   = hi * 8;
  const int n0   = wave * 16;
  const _Float16* BrowW = &sWhh[(n0 + mlo) * HP];
  const float bval = sBhh[n0 + mlo];

  // Stage xp tile for timestep tt into LDS buffer `buf` (async, ASYNCcnt).
  auto stage = [&](int tt, int buf) {
    const float* src = XP + ((size_t)tt * B + bt0) * N3;
#pragma unroll
    for (int q = 0; q < CPT; ++q) {
      int idx = tid + q * TPB;                       // float4 chunk index
      unsigned lo = (unsigned)(size_t)&sXp[buf][idx * 4];      // LDS byte offset
      unsigned long long ga = (unsigned long long)(const void*)(src + idx * 4);
      asm volatile("global_load_async_to_lds_b128 %0, %1, off"
                   :: "v"(lo), "v"(ga) : "memory");
    }
  };

  stage(0, 0);          // prologue: xp for t=0 in flight
  __syncthreads();      // sHh/sBhh ready

  for (int t = 0; t < S; ++t) {
    if (t + 1 < S) stage(t + 1, (t + 1) & 1);

    // hp tile for this wave: rows = batch, cols = [n0, n0+16)
    FragF c;
#pragma unroll
    for (int g = 0; g < 8; ++g) c.f[g] = bval;
#pragma unroll
    for (int kc = 0; kc < KCN; ++kc) {
      v16h a = ld_frag(&sHh[mlo * HP], kc * 32 + ko);
      v16h b = ld_frag(BrowW,          kc * 32 + ko);
      c.v = __builtin_amdgcn_wmma_f32_16x16x32_f16(false, a, false, b,
                                                   (short)0, c.v, false, false);
    }
#pragma unroll
    for (int g = 0; g < 8; ++g)
      sHp[(g + hi * 8) * N3 + n0 + mlo] = c.f[g];

    // Drain this step's staged xp (async loads complete in order, so allowing
    // the CPT just-issued t+1 loads to remain outstanding is safe).
    if (t + 1 < S) asm volatile("s_wait_asynccnt 0x2" ::: "memory");
    else           asm volatile("s_wait_asynccnt 0x0" ::: "memory");
    __syncthreads();

    // gate math: r,z,n ordering (PyTorch convention)
    const float* xp = sXp[t & 1];
    for (int e = tid; e < 16 * H; e += TPB) {
      int b = e / H, j = e % H;
      float r = sigm(xp[b * N3 + j]     + sHp[b * N3 + j]);
      float z = sigm(xp[b * N3 + H + j] + sHp[b * N3 + H + j]);
      float n = tanhf(xp[b * N3 + 2 * H + j] + r * sHp[b * N3 + 2 * H + j]);
      float hnew = (1.0f - z) * n + z * sH[e];
      sH[e] = hnew;
      sHh[b * HP + j] = (_Float16)hnew;
      if (Yout) {
        _Float16* yr = Yout + ((size_t)t * B + bt0 + b) * HP;
        yr[j] = (_Float16)hnew;
        if (HP > H) yr[H + j] = (_Float16)0.f;   // keep next layer's K-pad zero
      }
    }
    __syncthreads();
  }

  if (hT)
    for (int i = tid; i < 16 * H; i += TPB)
      hT[(size_t)(bt0 + i / H) * H + (i % H)] = sH[i];
}

// out = concat(h_last_layer, h_prev_layer) along feature dim -> (B, 2H)
__global__ void gather_out_kernel(const float* __restrict__ hLast,
                                  const float* __restrict__ hPrev,
                                  float* __restrict__ out, int B, int H) {
  int i = blockIdx.x * blockDim.x + threadIdx.x;
  if (i >= B * 2 * H) return;
  int b = i / (2 * H), j = i % (2 * H);
  out[i] = (j < H) ? hLast[(size_t)b * H + j] : hPrev[(size_t)b * H + (j - H)];
}

// ---------------------------------------------------------------------------
extern "C" void kernel_launch(void* const* d_in, const int* in_sizes, int n_in,
                              void* d_out, int out_size, void* d_ws, size_t ws_size,
                              hipStream_t stream) {
  (void)in_sizes; (void)n_in; (void)out_size; (void)ws_size;
  const int S = S_LEN, B = B_SZ, SB = S * B;

  struct Layer { int In, H, kpad, hp; const float *wih, *whh, *bih, *bhh; };
  const int Ins[7] = {300, 16, 16, 16, 16, 64, 64};
  const int Hs[7]  = { 16, 16, 16, 16, 64, 64, 64};
  Layer L[7];
  for (int l = 0; l < 7; ++l) {
    L[l].In = Ins[l]; L[l].H = Hs[l];
    L[l].kpad = (Ins[l] + 31) & ~31;
    L[l].hp   = (Hs[l] == 16) ? 32 : 64;
    L[l].wih = (const float*)d_in[1 + 4 * l + 0];
    L[l].whh = (const float*)d_in[1 + 4 * l + 1];
    L[l].bih = (const float*)d_in[1 + 4 * l + 2];
    L[l].bhh = (const float*)d_in[1 + 4 * l + 3];
  }
  const float* x = (const float*)d_in[0];

  // Workspace layout. X16 (f16 K-padded copy of x, SB x 320) is aliased into
  // the XP region at +32MB: layer 0 only writes XP rows with N=48 (< 25.2MB),
  // and X16 is dead before any later layer writes XP beyond that.
  char* ws = (char*)d_ws;
  float*    XP  = (float*)ws;
  _Float16* X16 = (_Float16*)(ws + ((size_t)32 << 20));
  size_t off = ((size_t)32 << 20) + (size_t)SB * 320 * sizeof(_Float16);  // ~112MB
  if (off < (size_t)SB * 192 * sizeof(float)) off = (size_t)SB * 192 * sizeof(float);
  auto alloc = [&](size_t bytes) -> void* {
    off = (off + 255) & ~(size_t)255;
    void* p = ws + off; off += bytes; return p;
  };

  _Float16* Y0 = (_Float16*)alloc((size_t)SB * 64 * sizeof(_Float16));  // ping
  _Float16* Y1 = (_Float16*)alloc((size_t)SB * 64 * sizeof(_Float16));  // pong
  _Float16 *wih16[7], *whh16[7];
  for (int l = 0; l < 7; ++l) {
    wih16[l] = (_Float16*)alloc((size_t)3 * L[l].H * L[l].kpad * sizeof(_Float16));
    whh16[l] = (_Float16*)alloc((size_t)3 * L[l].H * L[l].hp   * sizeof(_Float16));
  }
  float* h_c10 = (float*)alloc((size_t)B * 16 * sizeof(float));
  float* h_c11 = (float*)alloc((size_t)B * 16 * sizeof(float));
  float* h_d21 = (float*)alloc((size_t)B * 64 * sizeof(float));
  float* h_d22 = (float*)alloc((size_t)B * 64 * sizeof(float));

  // Convert weights and x to K-padded f16 (deterministic, every call).
  for (int l = 0; l < 7; ++l) {
    int rows = 3 * L[l].H;
    int n1 = rows * L[l].kpad;
    cvt_pad_kernel<<<(n1 + 255) / 256, 256, 0, stream>>>(L[l].wih, wih16[l], rows, L[l].In, L[l].kpad);
    int n2 = rows * L[l].hp;
    cvt_pad_kernel<<<(n2 + 255) / 256, 256, 0, stream>>>(L[l].whh, whh16[l], rows, L[l].H, L[l].hp);
  }
  {
    int nx = SB * 320;
    cvt_pad_kernel<<<(nx + 255) / 256, 256, 0, stream>>>(x, X16, SB, 300, 320);
  }

  const dim3 wv(32);
  // compress1 layer 0 (A = padded f16 x)
  xproj_kernel<3><<<dim3(SB / 16, 1), wv, 0, stream>>>(X16, 320, wih16[0], L[0].bih, XP,  48, SB);
  gru_scan_kernel<16><<<dim3(B / 16),  96, 0, stream>>>(XP, whh16[0], L[0].bhh, nullptr, Y0, h_c10, S, B);
  // compress1 layer 1
  xproj_kernel<3><<<dim3(SB / 16, 1), wv, 0, stream>>>(Y0,  32, wih16[1], L[1].bih, XP,  48, SB);
  gru_scan_kernel<16><<<dim3(B / 16),  96, 0, stream>>>(XP, whh16[1], L[1].bhh, nullptr, Y1, h_c11, S, B);
  // decode1 layer 0 (h0 = c1 layer0 final hidden)
  xproj_kernel<3><<<dim3(SB / 16, 1), wv, 0, stream>>>(Y1,  32, wih16[2], L[2].bih, XP,  48, SB);
  gru_scan_kernel<16><<<dim3(B / 16),  96, 0, stream>>>(XP, whh16[2], L[2].bhh, h_c10, Y0, nullptr, S, B);
  // decode1 layer 1 (h0 = c1 layer1 final hidden)
  xproj_kernel<3><<<dim3(SB / 16, 1), wv, 0, stream>>>(Y0,  32, wih16[3], L[3].bih, XP,  48, SB);
  gru_scan_kernel<16><<<dim3(B / 16),  96, 0, stream>>>(XP, whh16[3], L[3].bhh, h_c11, Y1, nullptr, S, B);
  // decode2 layer 0 (16 -> 64)
  xproj_kernel<3><<<dim3(SB / 16, 4), wv, 0, stream>>>(Y1,  32, wih16[4], L[4].bih, XP, 192, SB);
  gru_scan_kernel<64><<<dim3(B / 16), 384, 0, stream>>>(XP, whh16[4], L[4].bhh, nullptr, Y0, nullptr, S, B);
  // decode2 layer 1 (keep final hidden h3_1)
  xproj_kernel<3><<<dim3(SB / 16, 4), wv, 0, stream>>>(Y0,  64, wih16[5], L[5].bih, XP, 192, SB);
  gru_scan_kernel<64><<<dim3(B / 16), 384, 0, stream>>>(XP, whh16[5], L[5].bhh, nullptr, Y1, h_d21, S, B);
  // decode2 layer 2 (keep final hidden h3_2; y discarded)
  xproj_kernel<3><<<dim3(SB / 16, 4), wv, 0, stream>>>(Y1,  64, wih16[6], L[6].bih, XP, 192, SB);
  gru_scan_kernel<64><<<dim3(B / 16), 384, 0, stream>>>(XP, whh16[6], L[6].bhh, nullptr, (_Float16*)nullptr, h_d22, S, B);

  // out = concat(h3_2, h3_1) -> (B, 128) fp32
  gather_out_kernel<<<(B * 128 + 255) / 256, 256, 0, stream>>>(h_d22, h_d21, (float*)d_out, B, 64);
}
